// TripletLoss_9277129359404
// MI455X (gfx1250) — matrix operand order repaired and checked by reading, not compiled, not run on hardware
//
#include <hip/hip_runtime.h>
#include <hip/hip_bf16.h>

// CDNA5 (gfx1250) fused batch-hard triplet loss.
//  * Gram = X X^T via split-bf16 (hi/lo) V_WMMA_F32_16X16X32_BF16:
//    3 WMMAs per K=32 step -> ~fp32 accuracy for the mining comparisons.
//  * 2x2 register blocking per wave (32x32 macro-tile): 16 ds_load_b128 per
//    12 WMMAs.
//  * LDS staging via CDNA5 async global->LDS (GLOBAL_LOAD_ASYNC_TO_LDS_B128,
//    ASYNCcnt + s_wait_asynccnt) -- no VGPR round-trip for the bf16 tiles.
//  * Mining compares squared distances (sqrt is monotone) -> no sqrt in the
//    hot kernel; u64-packed (d2bits<<32 | j) argmax/argmin via shfl + LDS u64
//    atomics + one global u64 atomic per (row, block).
//  * One-shot fp32->bf16 hi/lo split pass (if workspace allows); exact fp32
//    recompute of chosen distances; fixed-order final reduction.

typedef __attribute__((ext_vector_type(16))) __bf16 v16bf;
typedef __attribute__((ext_vector_type(8)))  float  v8f;

#define KC       64    // K chunk staged in LDS
#define BM       64    // output tile rows per block
#define BN       128   // output tile cols per block
#define NROWS    (BM + BN)   // staged LDS rows (A then B)
#define LDSTRIDE 72    // padded LDS row stride in ushorts (16B aligned, conflict-free)

__device__ __forceinline__ unsigned short f2bf(float f) {
  unsigned u = __float_as_uint(f);
  unsigned r = u + 0x7FFFu + ((u >> 16) & 1u);   // round-to-nearest-even
  return (unsigned short)(r >> 16);
}
__device__ __forceinline__ float bf2f(unsigned short h) {
  return __uint_as_float(((unsigned)h) << 16);
}

// Async copy of 16B from global to LDS (GVS not needed; GV mode, ASYNCcnt).
__device__ __forceinline__ void async_g2l_b128(void* lds_dst, const void* gsrc) {
  const unsigned       ldsa = (unsigned)(unsigned long long)lds_dst; // low 32b = LDS offset
  const unsigned long long ga = (unsigned long long)gsrc;
  asm volatile("global_load_async_to_lds_b128 %0, %1, off"
               :
               : "v"(ldsa), "v"(ga)
               : "memory");
}
__device__ __forceinline__ void wait_async0() {
  asm volatile("s_wait_asynccnt 0" ::: "memory");
}

union Frag { v16bf v; uint4 q[2]; };

// ---------------------------------------------------------------- init / histo
__global__ void init_ws_kernel(unsigned long long* __restrict__ bp,
                               unsigned long long* __restrict__ bn,
                               int* __restrict__ counts, int n) {
  int i = blockIdx.x * blockDim.x + threadIdx.x;
  if (i < n) { bp[i] = 0ull; bn[i] = ~0ull; }
  if (i < 16) counts[i] = 0;
}

__global__ void label_hist_kernel(const int* __restrict__ labels,
                                  int* __restrict__ counts, int n) {
  int i = blockIdx.x * blockDim.x + threadIdx.x;
  if (i < n) atomicAdd(&counts[labels[i] & 15], 1);
}

// ------------------------------------------------- one-shot fp32 -> bf16 hi/lo
__global__ __launch_bounds__(256) void split_bf16_kernel(
    const float* __restrict__ X, unsigned short* __restrict__ Xhi,
    unsigned short* __restrict__ Xlo, int total4) {
  const int i = blockIdx.x * 256 + threadIdx.x;
  if (i >= total4) return;
  const float4 v = ((const float4*)X)[i];
  const float vv[4] = {v.x, v.y, v.z, v.w};
  unsigned h[4], l[4];
#pragma unroll
  for (int e = 0; e < 4; ++e) {
    h[e] = f2bf(vv[e]);
    l[e] = f2bf(vv[e] - bf2f((unsigned short)h[e]));
  }
  const uint2 ph = {h[0] | (h[1] << 16), h[2] | (h[3] << 16)};
  const uint2 pl = {l[0] | (l[1] << 16), l[2] | (l[3] << 16)};
  ((uint2*)Xhi)[i] = ph;
  ((uint2*)Xlo)[i] = pl;
}

// ---------------------------------------------------------------- row sum(x^2)
__global__ __launch_bounds__(256) void row_sumsq_kernel(
    const float* __restrict__ X, float* __restrict__ sq, int dim) {
  const int row = blockIdx.x;
  const int tid = threadIdx.x;
  float s = 0.f;
  for (int k = tid; k < dim; k += 256) {
    float v = X[(size_t)row * dim + k];
    s += v * v;
  }
#pragma unroll
  for (int m = 1; m < 32; m <<= 1) s += __shfl_xor(s, m, 32);
  __shared__ float ws[8];
  if ((tid & 31) == 0) ws[tid >> 5] = s;
  __syncthreads();
  if (tid == 0) {
    float t = 0.f;
#pragma unroll
    for (int q = 0; q < 8; ++q) t += ws[q];
    sq[row] = t;
  }
}

// ------------------------------------------- fused WMMA Gram + hardest mining
__global__ __launch_bounds__(256) void gram_mine_kernel(
    const float* __restrict__ X,
    const unsigned short* __restrict__ Xhi,
    const unsigned short* __restrict__ Xlo, int precomp,
    const float* __restrict__ sq, const int* __restrict__ labels,
    unsigned long long* __restrict__ bestPos,
    unsigned long long* __restrict__ bestNeg, int n, int dim) {
  __shared__ __align__(16) unsigned short sAhi[BM * LDSTRIDE];
  __shared__ __align__(16) unsigned short sAlo[BM * LDSTRIDE];
  __shared__ __align__(16) unsigned short sBhi[BN * LDSTRIDE];
  __shared__ __align__(16) unsigned short sBlo[BN * LDSTRIDE];
  __shared__ unsigned long long sBP[BM];
  __shared__ unsigned long long sBNg[BM];

  const int tid  = threadIdx.x;
  const int i0   = blockIdx.x * BM;
  const int j0   = blockIdx.y * BN;
  const int lane = tid & 31;
  const int w    = tid >> 5;    // wave 0..7
  const int wm   = w >> 2;      // 0..1  -> 32-row macro tile
  const int wn   = w & 3;       // 0..3  -> 32-col macro tile
  const int lr   = lane & 15;
  const int lh   = lane >> 4;

  if (tid < BM) { sBP[tid] = 0ull; sBNg[tid] = ~0ull; }

  v8f acc[2][2] = {};

  for (int kc = 0; kc < dim; kc += KC) {
    // ---- stage A rows [i0,i0+BM) and B rows [j0,j0+BN), cols [kc,kc+KC)
    if (precomp) {
      // async global->LDS copy of pre-split bf16 (16B per issue, ASYNCcnt)
#pragma unroll
      for (int it = 0; it < (NROWS * (KC / 8)) / 256; ++it) {   // 6 iters
        const int s   = tid + 256 * it;
        const int row = s >> 3;            // 0..191
        const int c8  = s & 7;             // 8-elem segment
        const int grow = (row < BM) ? (i0 + row) : (j0 + row - BM);
        const size_t go = (size_t)grow * dim + kc + c8 * 8;
        unsigned short* dh =
            ((row < BM) ? (sAhi + row * LDSTRIDE) : (sBhi + (row - BM) * LDSTRIDE)) + c8 * 8;
        unsigned short* dl =
            ((row < BM) ? (sAlo + row * LDSTRIDE) : (sBlo + (row - BM) * LDSTRIDE)) + c8 * 8;
        async_g2l_b128(dh, Xhi + go);
        async_g2l_b128(dl, Xlo + go);
      }
      wait_async0();
    } else {
      // fallback: split fp32 -> bf16 hi/lo on the fly
#pragma unroll
      for (int it = 0; it < (NROWS * (KC / 4)) / 256; ++it) {   // 12 iters
        const int s   = tid + 256 * it;
        const int row = s >> 4;            // 0..191
        const int c4  = s & 15;            // float4 segment
        const int grow = (row < BM) ? (i0 + row) : (j0 + row - BM);
        const float4 v = *(const float4*)(X + (size_t)grow * dim + kc + c4 * 4);
        const float vv[4] = {v.x, v.y, v.z, v.w};
        unsigned short* dh =
            ((row < BM) ? (sAhi + row * LDSTRIDE) : (sBhi + (row - BM) * LDSTRIDE)) + c4 * 4;
        unsigned short* dl =
            ((row < BM) ? (sAlo + row * LDSTRIDE) : (sBlo + (row - BM) * LDSTRIDE)) + c4 * 4;
#pragma unroll
        for (int e = 0; e < 4; ++e) {
          const unsigned short h = f2bf(vv[e]);
          dh[e] = h;
          dl[e] = f2bf(vv[e] - bf2f(h));
        }
      }
    }
    __syncthreads();

#pragma unroll
    for (int k32 = 0; k32 < KC; k32 += 32) {
      Frag ahi[2], alo[2], bhi[2], blo[2];
      const int ka = k32 + lh * 8;    // A: K = {8h..8h+7} then {16+8h..}
      const int kb = k32 + lh * 8;    // B: K = {8h..8h+7} then {16+8h..}
#pragma unroll
      for (int u = 0; u < 2; ++u) {
        const int ar = (wm * 32 + u * 16 + lr) * LDSTRIDE;
        ahi[u].q[0] = *(const uint4*)(sAhi + ar + ka);
        ahi[u].q[1] = *(const uint4*)(sAhi + ar + ka + 16);
        alo[u].q[0] = *(const uint4*)(sAlo + ar + ka);
        alo[u].q[1] = *(const uint4*)(sAlo + ar + ka + 16);
        const int br = (wn * 32 + u * 16 + lr) * LDSTRIDE;
        bhi[u].q[0] = *(const uint4*)(sBhi + br + kb);
        bhi[u].q[1] = *(const uint4*)(sBhi + br + kb + 16);
        blo[u].q[0] = *(const uint4*)(sBlo + br + kb);
        blo[u].q[1] = *(const uint4*)(sBlo + br + kb + 16);
      }
#pragma unroll
      for (int am = 0; am < 2; ++am) {
#pragma unroll
        for (int bn2 = 0; bn2 < 2; ++bn2) {
          acc[am][bn2] = __builtin_amdgcn_wmma_f32_16x16x32_bf16(
              false, ahi[am].v, false, bhi[bn2].v, (short)0, acc[am][bn2], false, false);
          acc[am][bn2] = __builtin_amdgcn_wmma_f32_16x16x32_bf16(
              false, ahi[am].v, false, blo[bn2].v, (short)0, acc[am][bn2], false, false);
          acc[am][bn2] = __builtin_amdgcn_wmma_f32_16x16x32_bf16(
              false, alo[am].v, false, bhi[bn2].v, (short)0, acc[am][bn2], false, false);
        }
      }
    }
    __syncthreads();
  }

  // Mining epilogue on SQUARED distances (sqrt monotone -> same arg winners).
  // C/D layout: VGPR t, lane l -> n = l%16, m = t + 8*(l/16).
#pragma unroll
  for (int ns = 0; ns < 2; ++ns) {
    const int   j    = j0 + wn * 32 + ns * 16 + lr;
    const float sqj  = sq[j];
    const int   labj = labels[j];
#pragma unroll
    for (int as2 = 0; as2 < 2; ++as2) {
#pragma unroll
      for (int t = 0; t < 8; ++t) {
        const int   m  = as2 * 16 + t + 8 * lh;
        const int   i  = i0 + wm * 32 + m;
        const float g  = acc[as2][ns][t];
        const float d2 = fmaxf(sq[i] + sqj - 2.0f * g, 0.0f);
        const bool same = (labels[i] == labj);
        const unsigned long long pk =
            (((unsigned long long)__float_as_uint(d2)) << 32) | (unsigned)j;
        unsigned long long pp = (same && (i != j)) ? pk : 0ull;  // max -> hardest pos
        unsigned long long pn = (!same) ? pk : ~0ull;            // min -> hardest neg
#pragma unroll
        for (int s = 1; s < 16; s <<= 1) {                       // 16 lanes share a row
          const unsigned long long op = __shfl_xor(pp, s, 32);
          const unsigned long long on = __shfl_xor(pn, s, 32);
          pp = (op > pp) ? op : pp;
          pn = (on < pn) ? on : pn;
        }
        if (lr == 0) {
          const int row = wm * 32 + m;
          atomicMax(&sBP[row], pp);
          atomicMin(&sBNg[row], pn);
        }
      }
    }
  }
  __syncthreads();
  if (tid < BM) {
    atomicMax(&bestPos[i0 + tid], sBP[tid]);
    atomicMin(&bestNeg[i0 + tid], sBNg[tid]);
  }
}

// --------------------------------------------- exact per-row triplet recompute
__global__ __launch_bounds__(256) void triplet_rows_kernel(
    const float* __restrict__ X, const int* __restrict__ labels,
    const unsigned long long* __restrict__ bestPos,
    const unsigned long long* __restrict__ bestNeg,
    const int* __restrict__ counts,
    float* __restrict__ perRow, int* __restrict__ validRow, int n, int dim) {
  const int tid  = threadIdx.x;
  const int lane = tid & 31;
  const int row  = blockIdx.x * 8 + (tid >> 5);
  if (row >= n) return;
  const int  c     = counts[labels[row] & 15];
  const bool valid = (c > 1) && (c < n);
  int pi = (int)(bestPos[row] & 0xFFFFFFFFull);
  int ni = (int)(bestNeg[row] & 0xFFFFFFFFull);
  if (pi < 0 || pi >= n) pi = row;
  if (ni < 0 || ni >= n) ni = row;
  const float* a = X + (size_t)row * dim;
  const float* p = X + (size_t)pi * dim;
  const float* q = X + (size_t)ni * dim;
  float sap = 0.f, san = 0.f;
  for (int k = lane; k < dim; k += 32) {
    const float av = a[k];
    const float dp = av - p[k] + 1e-6f;
    const float dn = av - q[k] + 1e-6f;
    sap += dp * dp;
    san += dn * dn;
  }
#pragma unroll
  for (int s = 1; s < 32; s <<= 1) {
    sap += __shfl_xor(sap, s, 32);
    san += __shfl_xor(san, s, 32);
  }
  if (lane == 0) {
    const float per = fmaxf(sqrtf(sap) - sqrtf(san) + 1.0f, 0.0f);
    perRow[row]   = valid ? per : 0.0f;
    validRow[row] = valid ? 1 : 0;
  }
}

// --------------------------------------------------- deterministic final mean
__global__ __launch_bounds__(1024) void final_reduce_kernel(
    const float* __restrict__ perRow, const int* __restrict__ validRow,
    float* __restrict__ out, int n) {
  __shared__ float sf[1024];
  __shared__ int   si[1024];
  const int tid = threadIdx.x;
  float s = 0.f;
  int   c = 0;
  for (int k = tid; k < n; k += 1024) { s += perRow[k]; c += validRow[k]; }
  sf[tid] = s;
  si[tid] = c;
  __syncthreads();
  for (int st = 512; st > 0; st >>= 1) {
    if (tid < st) { sf[tid] += sf[tid + st]; si[tid] += si[tid + st]; }
    __syncthreads();
  }
  if (tid == 0) out[0] = sf[0] / (float)(si[0] > 0 ? si[0] : 1);
}

// ----------------------------------------------------------------------- host
extern "C" void kernel_launch(void* const* d_in, const int* in_sizes, int n_in,
                              void* d_out, int out_size, void* d_ws, size_t ws_size,
                              hipStream_t stream) {
  (void)n_in; (void)out_size;
  const float* X      = (const float*)d_in[0];
  const int*   labels = (const int*)d_in[1];
  float*       out    = (float*)d_out;
  const int n   = in_sizes[1];       // 4096
  const int dim = in_sizes[0] / n;   // 512

  char* ws = (char*)d_ws;
  size_t off = 0;
  float* sq = (float*)(ws + off);                 off += (size_t)n * sizeof(float);
  off = (off + 255) & ~(size_t)255;
  unsigned long long* bestPos = (unsigned long long*)(ws + off); off += (size_t)n * 8;
  unsigned long long* bestNeg = (unsigned long long*)(ws + off); off += (size_t)n * 8;
  int*   counts   = (int*)(ws + off);             off += 256;
  float* perRow   = (float*)(ws + off);           off += (size_t)n * sizeof(float);
  int*   validRow = (int*)(ws + off);             off += (size_t)n * sizeof(int);
  off = (off + 255) & ~(size_t)255;
  unsigned short* Xhi = (unsigned short*)(ws + off); off += (size_t)n * dim * 2;
  unsigned short* Xlo = (unsigned short*)(ws + off); off += (size_t)n * dim * 2;
  const int precomp = (ws_size >= off) ? 1 : 0;

  const int blocks256 = (n + 255) / 256;
  init_ws_kernel<<<blocks256, 256, 0, stream>>>(bestPos, bestNeg, counts, n);
  label_hist_kernel<<<blocks256, 256, 0, stream>>>(labels, counts, n);
  row_sumsq_kernel<<<n, 256, 0, stream>>>(X, sq, dim);
  if (precomp) {
    const int total4 = (n * dim) / 4;
    split_bf16_kernel<<<(total4 + 255) / 256, 256, 0, stream>>>(X, Xhi, Xlo, total4);
  }
  dim3 g(n / BM, n / BN);
  gram_mine_kernel<<<g, 256, 0, stream>>>(X, Xhi, Xlo, precomp, sq, labels,
                                          bestPos, bestNeg, n, dim);
  triplet_rows_kernel<<<n / 8, 256, 0, stream>>>(X, labels, bestPos, bestNeg,
                                                 counts, perRow, validRow, n, dim);
  final_reduce_kernel<<<1, 1024, 0, stream>>>(perRow, validRow, out, n);
}
